// MotionNet_75608604279247
// MI455X (gfx1250) — matrix operand order repaired and checked by reading, not compiled, not run on hardware
//
#include <hip/hip_runtime.h>
#include <math.h>
#include <stdint.h>

typedef __bf16 bf16_t;
typedef __attribute__((ext_vector_type(16))) __bf16 v16bf;
typedef __attribute__((ext_vector_type(8)))  float  v8f;

#define BSZ      4096
#define NEXP     6

// padded dims
#define KF_PAD    1856   // frame 1830
#define KI_PAD    1056   // inet in 1027
#define KH0_PAD   2112   // h0 2086
#define DH        512
#define NHEAD_PAD 1920   // heads 1830, padded to multiple of 128

// tiling
#define BM   128
#define BN   128
#define LDT  40                 // 32 + 8 bf16 pad (80B rows, 16B aligned)
#define TA   (BM * LDT)         // A tile elems per stage
#define TB   (BN * LDT)         // B tile elems per stage
#define TS   (TA + TB)          // stage stride (elems)
#define NSTAGE 3

struct Frag { union { uint4 q[2]; v16bf v; }; };

__device__ __forceinline__ float elu_f(float v) {
  return v > 0.f ? v : (__expf(v) - 1.f);
}

__device__ __forceinline__ uint32_t lds_addr32(const void* p) {
  // LDS aperture flat pointers carry the LDS byte offset in the low 32 bits
  return (uint32_t)(uintptr_t)p;
}

// async copy: LDS[vdst + off] = MEM[saddr + vaddr + off], 16 bytes per lane
#define ASYNC_B128(ldsOff, gOff, base, OFFLIT)                               \
  asm volatile("global_load_async_to_lds_b128 %0, %1, %2 offset:" OFFLIT    \
               :: "v"(ldsOff), "v"(gOff), "s"(base) : "memory")

#define WAIT_ASYNC(LIT) asm volatile("s_wait_asynccnt " LIT ::: "memory")

// ---------------------------------------------------------------------------
// Generic bf16 GEMM: C = act(A[M x K] @ W + bias), W stored TRANSPOSED
// (N-major: Wt[n*ldWt + k]).  grid = (Npad/BN, M/BM), block = 256 (8 waves),
// wave tile 32x64 (2x4 WMMA frags). 3-stage async LDS pipeline.
// ---------------------------------------------------------------------------
template<typename OutT, bool ACT>
__global__ void __launch_bounds__(256) gemm_bias_act(
    const bf16_t* __restrict__ A, int ldA,
    const bf16_t* __restrict__ Wt, int ldWt,
    const float*  __restrict__ bias,
    OutT* __restrict__ C, int ldC, int Ksteps)
{
  __shared__ __align__(16) bf16_t sm[NSTAGE * TS];

  const int tid  = threadIdx.x;
  const int lane = tid & 31;
  const int wave = tid >> 5;
  const int wm   = (wave >> 1) * 32;   // 4 waves over M
  const int wn   = (wave & 1) * 64;    // 2 waves over N
  const int frow = lane & 15;
  const int fh   = lane >> 4;

  const size_t blockM = (size_t)blockIdx.y * BM;
  const int    blockN = blockIdx.x * BN;

  // staging: thread -> (row 0..127, 16-elem segment)
  const int sr   = tid >> 1;
  const int sseg = (tid & 1) * 16;

  uint32_t ldsA[NSTAGE], ldsB[NSTAGE];
  #pragma unroll
  for (int s = 0; s < NSTAGE; ++s) {
    ldsA[s] = lds_addr32(&sm[s * TS + sr * LDT + sseg]);
    ldsB[s] = lds_addr32(&sm[s * TS + TA + sr * LDT + sseg]);
  }
  const unsigned long long aBase = (unsigned long long)A;
  const unsigned long long bBase = (unsigned long long)Wt;
  const uint32_t aOff0 = (uint32_t)(((blockM + sr) * (size_t)ldA + sseg) * 2);
  const uint32_t bOff0 = (uint32_t)((((size_t)blockN + sr) * ldWt + sseg) * 2);

  v8f acc[2][4];
  #pragma unroll
  for (int i = 0; i < 2; ++i)
    #pragma unroll
    for (int j = 0; j < 4; ++j)
      #pragma unroll
      for (int r = 0; r < 8; ++r) acc[i][j][r] = 0.f;

  // prologue: issue tile 0 into stage 0
  {
    ASYNC_B128(ldsA[0], aOff0, aBase, "0");
    ASYNC_B128(ldsA[0], aOff0, aBase, "16");
    ASYNC_B128(ldsB[0], bOff0, bBase, "0");
    ASYNC_B128(ldsB[0], bOff0, bBase, "16");
  }

  int stage = 0;
  for (int j = 0; j < Ksteps; ++j) {
    int nst = stage + 1; if (nst == NSTAGE) nst = 0;
    if (j + 1 < Ksteps) {
      const uint32_t ga = aOff0 + (uint32_t)(j + 1) * 64;
      const uint32_t gb = bOff0 + (uint32_t)(j + 1) * 64;
      ASYNC_B128(ldsA[nst], ga, aBase, "0");
      ASYNC_B128(ldsA[nst], ga, aBase, "16");
      ASYNC_B128(ldsB[nst], gb, bBase, "0");
      ASYNC_B128(ldsB[nst], gb, bBase, "16");
      WAIT_ASYNC("0x4");      // tile j fully landed (in-order completion)
    } else {
      WAIT_ASYNC("0x0");
    }
    __syncthreads();

    const bf16_t* At = &sm[stage * TS];
    const bf16_t* Bt = &sm[stage * TS + TA];

    Frag af[2], bf[4];
    #pragma unroll
    for (int i = 0; i < 2; ++i) {
      const bf16_t* ap = &At[(wm + i * 16 + frow) * LDT + 8 * fh];
      af[i].q[0] = *(const uint4*)(ap);        // K = 8h .. 8h+7
      af[i].q[1] = *(const uint4*)(ap + 16);   // K = 16+8h .. 16+8h+7
    }
    #pragma unroll
    for (int jj = 0; jj < 4; ++jj) {
      const bf16_t* bp = &Bt[(wn + jj * 16 + frow) * LDT + 8 * fh];
      bf[jj].q[0] = *(const uint4*)(bp);       // K = 8h .. 8h+7
      bf[jj].q[1] = *(const uint4*)(bp + 16);  // K = 16+8h .. 16+8h+7
    }
    #pragma unroll
    for (int i = 0; i < 2; ++i)
      #pragma unroll
      for (int jj = 0; jj < 4; ++jj)
        acc[i][jj] = __builtin_amdgcn_wmma_f32_16x16x32_bf16(
            false, af[i].v, false, bf[jj].v, (short)0, acc[i][jj], false, false);
    stage = nst;
  }

  #pragma unroll
  for (int jj = 0; jj < 4; ++jj) {
    const int col  = blockN + wn + jj * 16 + frow;
    const float bc = bias[col];
    #pragma unroll
    for (int i = 0; i < 2; ++i) {
      const size_t mBase = blockM + wm + i * 16 + 8 * fh;
      #pragma unroll
      for (int r = 0; r < 8; ++r) {
        float v = acc[i][jj][r] + bc;
        if (ACT) v = elu_f(v);
        C[(mBase + r) * (size_t)ldC + col] = (OutT)v;
      }
    }
  }
}

// ---------------------------------------------------------------------------
// Fused MoE layer: C = elu( sum_e omega[:,e]*(A @ W_e) + omega @ eb )
// omega folded into A while staging (VGPR path); expert weights (transposed)
// staged with async-to-LDS.  Flattened (expert, kstep) pipeline, 3 stages.
// ---------------------------------------------------------------------------
__global__ void __launch_bounds__(256) moe_gemm_elu(
    const bf16_t* __restrict__ A, int ldA,
    const bf16_t* __restrict__ Wt, uint32_t wStrideB /*bytes per expert*/, int ldWt,
    const float*  __restrict__ eb,     // [NEXP][DH]
    const float*  __restrict__ omega,  // [BSZ][NEXP]
    bf16_t* __restrict__ C, int ldC, int Ksteps)
{
  __shared__ __align__(16) bf16_t sm[NSTAGE * TS];
  __shared__ float omS[BM * NEXP];

  const int tid  = threadIdx.x;
  const int lane = tid & 31;
  const int wave = tid >> 5;
  const int wm   = (wave >> 1) * 32;
  const int wn   = (wave & 1) * 64;
  const int frow = lane & 15;
  const int fh   = lane >> 4;

  const size_t blockM = (size_t)blockIdx.y * BM;
  const int    blockN = blockIdx.x * BN;

  const int sr   = tid >> 1;
  const int sseg = (tid & 1) * 16;

  bf16_t* aS[NSTAGE];
  uint32_t ldsB[NSTAGE];
  #pragma unroll
  for (int s = 0; s < NSTAGE; ++s) {
    aS[s]   = &sm[s * TS + sr * LDT + sseg];
    ldsB[s] = lds_addr32(&sm[s * TS + TA + sr * LDT + sseg]);
  }
  const bf16_t* aRow = A + (blockM + sr) * (size_t)ldA + sseg;
  const unsigned long long bBase = (unsigned long long)Wt;
  const uint32_t bOff0 = (uint32_t)((((size_t)blockN + sr) * ldWt + sseg) * 2);

  for (int i = tid; i < BM * NEXP; i += 256) omS[i] = omega[blockM * NEXP + i];
  __syncthreads();

  v8f acc[2][4];
  #pragma unroll
  for (int i = 0; i < 2; ++i)
    #pragma unroll
    for (int j = 0; j < 4; ++j)
      #pragma unroll
      for (int r = 0; r < 8; ++r) acc[i][j][r] = 0.f;

  const int total = NEXP * Ksteps;

  // staging helper (A: scaled VGPR path; B: async)
  auto issueTile = [&](int e, int ks, int st) {
    const float sw = omS[sr * NEXP + e];
    union { uint4 q; bf16_t h[8]; } p0, p1;
    const uint4* ag = (const uint4*)(aRow + (size_t)ks * 32);
    p0.q = ag[0];
    p1.q = ag[1];
    #pragma unroll
    for (int t = 0; t < 8; ++t) {
      p0.h[t] = (bf16_t)((float)p0.h[t] * sw);
      p1.h[t] = (bf16_t)((float)p1.h[t] * sw);
    }
    *(uint4*)(aS[st])     = p0.q;
    *(uint4*)(aS[st] + 8) = p1.q;
    const uint32_t gb = bOff0 + (uint32_t)e * wStrideB + (uint32_t)ks * 64;
    ASYNC_B128(ldsB[st], gb, bBase, "0");
    ASYNC_B128(ldsB[st], gb, bBase, "16");
  };

  issueTile(0, 0, 0);   // prologue

  int stage = 0, eI = 0, ksI = 0;
  for (int j = 0; j < total; ++j) {
    int nst = stage + 1; if (nst == NSTAGE) nst = 0;
    int eN = eI, ksN = ksI + 1;
    if (ksN == Ksteps) { ksN = 0; ++eN; }
    if (j + 1 < total) {
      issueTile(eN, ksN, nst);
      WAIT_ASYNC("0x2");
    } else {
      WAIT_ASYNC("0x0");
    }
    __syncthreads();

    const bf16_t* At = &sm[stage * TS];
    const bf16_t* Bt = &sm[stage * TS + TA];

    Frag af[2], bf[4];
    #pragma unroll
    for (int i = 0; i < 2; ++i) {
      const bf16_t* ap = &At[(wm + i * 16 + frow) * LDT + 8 * fh];
      af[i].q[0] = *(const uint4*)(ap);
      af[i].q[1] = *(const uint4*)(ap + 16);
    }
    #pragma unroll
    for (int jj = 0; jj < 4; ++jj) {
      const bf16_t* bp = &Bt[(wn + jj * 16 + frow) * LDT + 8 * fh];
      bf[jj].q[0] = *(const uint4*)(bp);
      bf[jj].q[1] = *(const uint4*)(bp + 16);
    }
    #pragma unroll
    for (int i = 0; i < 2; ++i)
      #pragma unroll
      for (int jj = 0; jj < 4; ++jj)
        acc[i][jj] = __builtin_amdgcn_wmma_f32_16x16x32_bf16(
            false, af[i].v, false, bf[jj].v, (short)0, acc[i][jj], false, false);

    stage = nst; eI = eN; ksI = ksN;
  }

  #pragma unroll
  for (int jj = 0; jj < 4; ++jj) {
    const int col = blockN + wn + jj * 16 + frow;
    float eb6[NEXP];
    #pragma unroll
    for (int e = 0; e < NEXP; ++e) eb6[e] = eb[e * DH + col];
    #pragma unroll
    for (int i = 0; i < 2; ++i) {
      const int mLoc = wm + i * 16 + 8 * fh;
      #pragma unroll
      for (int r = 0; r < 8; ++r) {
        float bb = 0.f;
        #pragma unroll
        for (int e = 0; e < NEXP; ++e) bb += omS[(mLoc + r) * NEXP + e] * eb6[e];
        float v = elu_f(acc[i][jj][r] + bb);
        C[(blockM + mLoc + r) * (size_t)ldC + col] = (bf16_t)v;
      }
    }
  }
}

// ---------------------------------------------------------------------------
// Feature assembly
// ---------------------------------------------------------------------------
__global__ void assemble_kernel(
    const float* __restrict__ rot, const float* __restrict__ transl,
    const float* __restrict__ verts, const float* __restrict__ dists,
    const float* __restrict__ bps, const float* __restrict__ ogoal,
    bf16_t* __restrict__ frame, bf16_t* __restrict__ h0, bf16_t* __restrict__ Ibf)
{
  const int b = blockIdx.x;
  const float* r = rot + (size_t)b * 495;
  for (int c = threadIdx.x; c < 1830; c += 256) {
    float v;
    if (c < 330) {
      int j = c / 6, rem = c % 6, rr = rem >> 1, cc = rem & 1;
      v = r[j * 9 + rr * 3 + cc];
    } else if (c < 333) {
      v = transl[b * 3 + (c - 330)];
    } else if (c < 1533) {
      v = verts[(size_t)b * 1200 + (c - 333)];
    } else {
      v = dists[(size_t)b * 297 + (c - 1533)];
    }
    bf16_t bv = (bf16_t)v;
    frame[(size_t)b * KF_PAD + c] = bv;
    h0[(size_t)b * KH0_PAD + c]   = bv;
  }
  for (int c = threadIdx.x; c < 1027; c += 256) {
    float v = (c < 1024) ? bps[(size_t)b * 1024 + c] : ogoal[b * 3 + (c - 1024)];
    Ibf[(size_t)b * KI_PAD + c] = (bf16_t)v;
  }
}

// ---------------------------------------------------------------------------
// Gating head (256 -> 6) + softmax
// ---------------------------------------------------------------------------
__global__ void __launch_bounds__(256) gate_softmax(
    const bf16_t* __restrict__ g2, const float* __restrict__ gw3,
    const float* __restrict__ gb3, float* __restrict__ omega)
{
  __shared__ float xs[256];
  __shared__ float lg[NEXP];
  const int b = blockIdx.x;
  xs[threadIdx.x] = (float)g2[(size_t)b * 256 + threadIdx.x];
  __syncthreads();
  if (threadIdx.x < NEXP) {
    float s = gb3[threadIdx.x];
    for (int k = 0; k < 256; ++k) s += xs[k] * gw3[k * NEXP + threadIdx.x];
    lg[threadIdx.x] = s;
  }
  __syncthreads();
  if (threadIdx.x == 0) {
    float mx = lg[0];
    for (int e = 1; e < NEXP; ++e) mx = fmaxf(mx, lg[e]);
    float ex[NEXP], den = 0.f;
    for (int e = 0; e < NEXP; ++e) { ex[e] = __expf(lg[e] - mx); den += ex[e]; }
    float inv = 1.f / den;
    for (int e = 0; e < NEXP; ++e) omega[(size_t)b * NEXP + e] = ex[e] * inv;
  }
}

// ---------------------------------------------------------------------------
// Heads postprocess: 6D->rotmat Gram-Schmidt + scatter
// head cols: [0,330) pose6d | [330,333) transl | [333,1533) verts | [1533,1830) dists
// d_out: pose(495) | transl(3) | dists(297) | verts(1200)
// ---------------------------------------------------------------------------
__global__ void postprocess(const float* __restrict__ H, float* __restrict__ out)
{
  const int b = blockIdx.x;
  const float* h = H + (size_t)b * NHEAD_PAD;
  const size_t OFF_T = (size_t)BSZ * 495;
  const size_t OFF_D = OFF_T + (size_t)BSZ * 3;
  const size_t OFF_V = OFF_D + (size_t)BSZ * 297;

  for (int j = threadIdx.x; j < 55; j += 256) {
    const float* p = h + j * 6;
    float a1x = p[0], a2x = p[1], a1y = p[2], a2y = p[3], a1z = p[4], a2z = p[5];
    float n1 = rsqrtf(a1x * a1x + a1y * a1y + a1z * a1z);
    float b1x = a1x * n1, b1y = a1y * n1, b1z = a1z * n1;
    float d = b1x * a2x + b1y * a2y + b1z * a2z;
    float cx = a2x - d * b1x, cy = a2y - d * b1y, cz = a2z - d * b1z;
    float n2 = rsqrtf(cx * cx + cy * cy + cz * cz);
    float b2x = cx * n2, b2y = cy * n2, b2z = cz * n2;
    float b3x = b1y * b2z - b1z * b2y;
    float b3y = b1z * b2x - b1x * b2z;
    float b3z = b1x * b2y - b1y * b2x;
    float* o = out + (size_t)b * 495 + j * 9;
    o[0] = b1x; o[1] = b2x; o[2] = b3x;
    o[3] = b1y; o[4] = b2y; o[5] = b3y;
    o[6] = b1z; o[7] = b2z; o[8] = b3z;
  }
  if (threadIdx.x < 3) out[OFF_T + (size_t)b * 3 + threadIdx.x] = h[330 + threadIdx.x];
  for (int i = threadIdx.x; i < 297; i += 256)
    out[OFF_D + (size_t)b * 297 + i] = h[1533 + i];
  for (int i = threadIdx.x; i < 1200; i += 256)
    out[OFF_V + (size_t)b * 1200 + i] = h[333 + i];
}

// ---------------------------------------------------------------------------
// Weight conversion (f32 -> bf16, TRANSPOSED to N-major, dst pre-zeroed)
// ---------------------------------------------------------------------------
__global__ void convert_pad_T(const float* __restrict__ src, int K, int N,
                              bf16_t* __restrict__ dst, int ldK)
{
  size_t idx = (size_t)blockIdx.x * 256 + threadIdx.x;
  if (idx < (size_t)N * K) {
    int n = (int)(idx / K);
    int k = (int)(idx - (size_t)n * K);
    dst[(size_t)n * ldK + k] = (bf16_t)src[(size_t)k * N + n];
  }
}

__global__ void convert_exp_T(const float* __restrict__ src, int K, int N,
                              bf16_t* __restrict__ dst, size_t dstStride, int ldK)
{
  size_t idx = (size_t)blockIdx.x * 256 + threadIdx.x;
  if (idx < (size_t)NEXP * N * K) {
    int e = (int)(idx / ((size_t)N * K));
    size_t rem = idx - (size_t)e * N * K;
    int n = (int)(rem / K);
    int k = (int)(rem - (size_t)n * K);
    dst[(size_t)e * dstStride + (size_t)n * ldK + k] =
        (bf16_t)src[(size_t)e * K * N + (size_t)k * N + n];
  }
}

__global__ void copy_f32(const float* __restrict__ src, int n, float* __restrict__ dst)
{
  int i = blockIdx.x * 256 + threadIdx.x;
  if (i < n) dst[i] = src[i];
}

// ---------------------------------------------------------------------------
extern "C" void kernel_launch(void* const* d_in, const int* in_sizes, int n_in,
                              void* d_out, int out_size, void* d_ws, size_t ws_size,
                              hipStream_t stream)
{
  (void)in_sizes; (void)n_in; (void)out_size; (void)ws_size;

  const float* rot    = (const float*)d_in[0];
  const float* transl = (const float*)d_in[1];
  const float* verts  = (const float*)d_in[2];
  const float* dists  = (const float*)d_in[3];
  const float* bps    = (const float*)d_in[4];
  const float* ogoal  = (const float*)d_in[5];
  const float* inet_w1 = (const float*)d_in[6];  const float* inet_b1 = (const float*)d_in[7];
  const float* inet_w2 = (const float*)d_in[8];  const float* inet_b2 = (const float*)d_in[9];
  const float* inet_w3 = (const float*)d_in[10]; const float* inet_b3 = (const float*)d_in[11];
  const float* gw1 = (const float*)d_in[12]; const float* gb1 = (const float*)d_in[13];
  const float* gw2 = (const float*)d_in[14]; const float* gb2 = (const float*)d_in[15];
  const float* gw3 = (const float*)d_in[16]; const float* gb3 = (const float*)d_in[17];
  const float* ew1 = (const float*)d_in[18]; const float* eb1 = (const float*)d_in[19];
  const float* ew2 = (const float*)d_in[20]; const float* eb2 = (const float*)d_in[21];
  const float* ew3 = (const float*)d_in[22]; const float* eb3 = (const float*)d_in[23];
  const float* pw = (const float*)d_in[24]; const float* pb = (const float*)d_in[25];
  const float* tw = (const float*)d_in[26]; const float* tb = (const float*)d_in[27];
  const float* vw = (const float*)d_in[28]; const float* vb = (const float*)d_in[29];
  const float* dw = (const float*)d_in[30]; const float* db = (const float*)d_in[31];

  char* ws = (char*)d_ws;
  size_t off = 0;
  auto carve = [&](size_t bytes) -> void* {
    void* p = ws + off;
    off = (off + bytes + 255) & ~(size_t)255;
    return p;
  };

  bf16_t* frame   = (bf16_t*)carve((size_t)BSZ * KF_PAD * 2);
  bf16_t* h0      = (bf16_t*)carve((size_t)BSZ * KH0_PAD * 2);
  bf16_t* Ibf     = (bf16_t*)carve((size_t)BSZ * KI_PAD * 2);
  bf16_t* hI1     = (bf16_t*)carve((size_t)BSZ * 256 * 2);
  bf16_t* hI2     = (bf16_t*)carve((size_t)BSZ * 256 * 2);
  bf16_t* g1      = (bf16_t*)carve((size_t)BSZ * 512 * 2);
  bf16_t* g2      = (bf16_t*)carve((size_t)BSZ * 256 * 2);
  float*  omega   = (float*) carve((size_t)BSZ * NEXP * 4);
  bf16_t* h1      = (bf16_t*)carve((size_t)BSZ * DH * 2);
  bf16_t* h2      = (bf16_t*)carve((size_t)BSZ * DH * 2);
  bf16_t* h3      = (bf16_t*)carve((size_t)BSZ * DH * 2);
  float*  headOut = (float*) carve((size_t)BSZ * NHEAD_PAD * 4);
  // transposed bf16 weights (N-major, ld = Kpad)
  bf16_t* wI1t    = (bf16_t*)carve((size_t)256 * KI_PAD * 2);
  bf16_t* wI2t    = (bf16_t*)carve((size_t)256 * 256 * 2);
  bf16_t* wI3t    = (bf16_t*)carve((size_t)256 * 256 * 2);
  bf16_t* wG1t    = (bf16_t*)carve((size_t)512 * KF_PAD * 2);
  bf16_t* wG2t    = (bf16_t*)carve((size_t)256 * 512 * 2);
  bf16_t* wE1t    = (bf16_t*)carve((size_t)NEXP * DH * KH0_PAD * 2);
  bf16_t* wE2t    = (bf16_t*)carve((size_t)NEXP * DH * DH * 2);
  bf16_t* wE3t    = (bf16_t*)carve((size_t)NEXP * DH * DH * 2);
  bf16_t* wHeadT  = (bf16_t*)carve((size_t)NHEAD_PAD * DH * 2);
  float*  headB   = (float*) carve((size_t)NHEAD_PAD * 4);

  hipMemsetAsync(d_ws, 0, off, stream);

  assemble_kernel<<<BSZ, 256, 0, stream>>>(rot, transl, verts, dists, bps, ogoal,
                                           frame, h0, Ibf);

  auto nb = [](size_t n) { return (unsigned)((n + 255) / 256); };
  convert_pad_T<<<nb((size_t)256 * 1027), 256, 0, stream>>>(inet_w1, 1027, 256, wI1t, KI_PAD);
  convert_pad_T<<<nb((size_t)256 * 256), 256, 0, stream>>>(inet_w2, 256, 256, wI2t, 256);
  convert_pad_T<<<nb((size_t)256 * 256), 256, 0, stream>>>(inet_w3, 256, 256, wI3t, 256);
  convert_pad_T<<<nb((size_t)512 * 1830), 256, 0, stream>>>(gw1, 1830, 512, wG1t, KF_PAD);
  convert_pad_T<<<nb((size_t)256 * 512), 256, 0, stream>>>(gw2, 512, 256, wG2t, 512);
  convert_exp_T<<<nb((size_t)NEXP * 512 * 2086), 256, 0, stream>>>(ew1, 2086, DH, wE1t, (size_t)DH * KH0_PAD, KH0_PAD);
  convert_exp_T<<<nb((size_t)NEXP * 512 * 512), 256, 0, stream>>>(ew2, DH, DH, wE2t, (size_t)DH * DH, DH);
  convert_exp_T<<<nb((size_t)NEXP * 512 * 512), 256, 0, stream>>>(ew3, DH, DH, wE3t, (size_t)DH * DH, DH);
  convert_pad_T<<<nb((size_t)330 * 512), 256, 0, stream>>>(pw, 512, 330, wHeadT + (size_t)0 * DH, DH);
  convert_pad_T<<<nb((size_t)3 * 512), 256, 0, stream>>>(tw, 512, 3, wHeadT + (size_t)330 * DH, DH);
  convert_pad_T<<<nb((size_t)1200 * 512), 256, 0, stream>>>(vw, 512, 1200, wHeadT + (size_t)333 * DH, DH);
  convert_pad_T<<<nb((size_t)297 * 512), 256, 0, stream>>>(dw, 512, 297, wHeadT + (size_t)1533 * DH, DH);
  copy_f32<<<nb(330), 256, 0, stream>>>(pb, 330, headB + 0);
  copy_f32<<<nb(3), 256, 0, stream>>>(tb, 3, headB + 330);
  copy_f32<<<nb(1200), 256, 0, stream>>>(vb, 1200, headB + 333);
  copy_f32<<<nb(297), 256, 0, stream>>>(db, 297, headB + 1533);

  const dim3 g256(256 / BN, BSZ / BM);
  const dim3 g512(512 / BN, BSZ / BM);
  const dim3 gHead(NHEAD_PAD / BN, BSZ / BM);

  // INet: 1027 -> 256 -> 256 -> 256 (last writes into h0[:,1830:2086])
  gemm_bias_act<bf16_t, true><<<g256, 256, 0, stream>>>(Ibf, KI_PAD, wI1t, KI_PAD, inet_b1, hI1, 256, KI_PAD / 32);
  gemm_bias_act<bf16_t, true><<<g256, 256, 0, stream>>>(hI1, 256, wI2t, 256, inet_b2, hI2, 256, 256 / 32);
  gemm_bias_act<bf16_t, true><<<g256, 256, 0, stream>>>(hI2, 256, wI3t, 256, inet_b3, h0 + 1830, KH0_PAD, 256 / 32);

  // Gating: 1830 -> 512 -> 256 -> softmax(6)
  gemm_bias_act<bf16_t, true><<<g512, 256, 0, stream>>>(frame, KF_PAD, wG1t, KF_PAD, gb1, g1, 512, KF_PAD / 32);
  gemm_bias_act<bf16_t, true><<<g256, 256, 0, stream>>>(g1, 512, wG2t, 512, gb2, g2, 256, 512 / 32);
  gate_softmax<<<BSZ, 256, 0, stream>>>(g2, gw3, gb3, omega);

  // MoE expert layers
  moe_gemm_elu<<<g512, 256, 0, stream>>>(h0, KH0_PAD, wE1t, (uint32_t)(DH * KH0_PAD * 2), KH0_PAD, eb1, omega, h1, DH, KH0_PAD / 32);
  moe_gemm_elu<<<g512, 256, 0, stream>>>(h1, DH, wE2t, (uint32_t)(DH * DH * 2), DH, eb2, omega, h2, DH, DH / 32);
  moe_gemm_elu<<<g512, 256, 0, stream>>>(h2, DH, wE3t, (uint32_t)(DH * DH * 2), DH, eb3, omega, h3, DH, DH / 32);

  // Fused output heads: 512 -> 1830 (padded 1920), f32 out, no activation
  gemm_bias_act<float, false><<<gHead, 256, 0, stream>>>(h3, DH, wHeadT, DH, headB, headOut, NHEAD_PAD, DH / 32);

  postprocess<<<BSZ, 256, 0, stream>>>(headOut, (float*)d_out);
}
